// GraphCastNet_38147899523434
// MI455X (gfx1250) — matrix operand order repaired and checked by reading, not compiled, not run on hardware
//
#include <hip/hip_runtime.h>
#include <hip/hip_bf16.h>
#include <cstdint>
#include <cstddef>

// ---------------------------------------------------------------------------
// GraphCast forward on gfx1250 using v_wmma_f32_16x16x32_f16.
// Fused MLP kernel:  Y = [LN]( SiLU(X@W1 + b1) @ W2 + b2 ) [+resid]
// Weight chunks are staged with GLOBAL_LOAD_ASYNC_TO_LDS_B128 (ASYNCcnt) --
// no VGPR staging, copies for chunk k+1 overlap the WMMA chain of chunk k.
// X gather-concat staged via registers (needs masking + f32->f16 convert).
// ---------------------------------------------------------------------------

typedef _Float16 v16h __attribute__((ext_vector_type(16)));
typedef float    v8f  __attribute__((ext_vector_type(8)));
typedef int      v4i128 __attribute__((vector_size(16)));   // 16B block for async copies
typedef __attribute__((address_space(1))) v4i128* gptr128;
typedef __attribute__((address_space(3))) v4i128* lptr128;

#define BM    64            // rows per workgroup tile
#define XSTR  34            // LDS stride (halves) for X chunk (64 x 32)
#define BSTR  40            // LDS stride (halves) for W chunk (256 x 32), 16B rows
#define HSTR  264           // LDS stride (halves) for H (64 x 256)
#define XBYTES (BM * XSTR * 2)            // 4352
#define BBYTES (256 * BSTR * 2)           // 20480
#define SETBYTES (XBYTES + BBYTES)        // 24832
#define REGION_BYTES 66560                // >= max(2*SETBYTES=49664, Ys 64*260*4)

#if __has_builtin(__builtin_amdgcn_global_load_async_to_lds_b128)
#define HAVE_ASYNC 1
#else
#define HAVE_ASYNC 0
#endif

union Frag16 { v16h v; _Float16 h[16]; unsigned u[8]; };
union PkH2   { _Float16 h[2]; unsigned u; };

__device__ __forceinline__ v8f wmma16(const Frag16& a, const Frag16& b, v8f c) {
    return __builtin_amdgcn_wmma_f32_16x16x32_f16(false, a.v, false, b.v,
                                                  (short)0, c, false, false);
}

__device__ __forceinline__ void wait_async() {
#if HAVE_ASYNC
#if __has_builtin(__builtin_amdgcn_s_wait_asynccnt)
    __builtin_amdgcn_s_wait_asynccnt(0);
#else
    asm volatile("s_wait_asynccnt 0x0" ::: "memory");
#endif
#endif
}

// fast sigmoid: single v_rcp_f32 instead of the IEEE division sequence
__device__ __forceinline__ float fast_sigmoid(float x) {
#if __has_builtin(__builtin_amdgcn_rcpf)
    return __builtin_amdgcn_rcpf(1.f + __expf(-x));
#else
    return 1.f / (1.f + __expf(-x));
#endif
}

// Convert fp32 weight [K][N] -> transposed f16 [N][Kp] (K padded to Kp, zero fill)
__global__ __launch_bounds__(256)
void k_cvt_t(const float* __restrict__ in, _Float16* __restrict__ out,
             int K, int N, int Kp) {
    int i = blockIdx.x * 256 + threadIdx.x;
    if (i < N * Kp) {
        int n = i / Kp, k = i - n * Kp;
        out[i] = (k < K) ? (_Float16)in[(size_t)k * N + n] : (_Float16)0.f;
    }
}

template <int DOUT>
__global__ __launch_bounds__(256)
void k_mlp(int M, int din,
           const float* __restrict__ s0, const int* __restrict__ i0, int w0,
           const float* __restrict__ s1, const int* __restrict__ i1,
           const float* __restrict__ s2, const int* __restrict__ i2,
           const _Float16* __restrict__ w1t, const float* __restrict__ b1,
           const _Float16* __restrict__ w2t, const float* __restrict__ b2,
           const float* __restrict__ gw, const float* __restrict__ bw,   // LN (null => none)
           const float* __restrict__ resid,                              // null or [M,DOUT]
           float* __restrict__ outp,                                     // null => no store
           float* __restrict__ aggp, const int* __restrict__ dsti,       // scatter-add
           const float* __restrict__ fstd, const float* __restrict__ fmean,
           const float* __restrict__ fgnf)                               // final head
{
    __shared__ __align__(16) _Float16 hs[BM * HSTR];
    __shared__ __align__(16) char region[REGION_BYTES];

    // runtime pointer arithmetic only (avoids addrspacecast-in-initializer bug)
    auto xsbuf = [&](int i) -> _Float16* {
        return (_Float16*)(region + i * SETBYTES);
    };
    auto bsbuf = [&](int i) -> _Float16* {
        return (_Float16*)(region + i * SETBYTES + XBYTES);
    };
    auto b2buf = [&](int i) -> _Float16* {
        return (_Float16*)(region + i * BBYTES);
    };

    const int tid  = threadIdx.x;
    const int lane = tid & 31;
    const int wav  = tid >> 5;
    const int rowTile = wav & 3;       // 4 row tiles of 16
    const int colHalf = wav >> 2;      // 2 col halves
    const int m     = lane & 15;
    const int khalf = lane >> 4;
    const long R0   = (long)blockIdx.x * BM;
    const int dinp  = (din + 31) & ~31;
    const int nk1   = (din + 31) >> 5;
    const int srow  = tid >> 2;        // staging row 0..63
    const int skb   = (tid & 3) * 8;   // staging k offset

    float xreg[8];

    auto loadX = [&](int kc) {
        const int k0 = kc << 5;
        const long R = R0 + srow;
        const int kstart = k0 + skb;
#pragma unroll
        for (int j = 0; j < 8; ++j) xreg[j] = 0.f;
        if (R < M && kstart < din) {
            const float* sp; const int* ip; int w, cb;
            if (kstart < w0)            { sp = s0; ip = i0; w = w0;  cb = kstart; }
            else if (kstart < w0 + 256) { sp = s1; ip = i1; w = 256; cb = kstart - w0; }
            else                        { sp = s2; ip = i2; w = 256; cb = kstart - w0 - 256; }
            long rr = ip ? (long)ip[R] : R;
            const float* base = sp + rr * (long)w + cb;
#pragma unroll
            for (int j = 0; j < 8; ++j)
                if (kstart + j < din) xreg[j] = base[j];
        }
    };
    auto storeX = [&](_Float16* xbuf) {
        _Float16* xp = &xbuf[srow * XSTR + skb];
#pragma unroll
        for (int t = 0; t < 4; ++t) {
            PkH2 pk;
            pk.h[0] = (_Float16)xreg[2 * t];
            pk.h[1] = (_Float16)xreg[2 * t + 1];
            *(unsigned*)(xp + 2 * t) = pk.u;
        }
    };

    // ---- weight chunk staging: async global->LDS (no VGPRs) ----
    auto stageW = [&](const _Float16* wsrc, size_t rowStride, int kc, _Float16* bbuf,
                      bool active) {
        char* g = (char*)(wsrc + (size_t)tid * rowStride + (kc << 5));
        char* l = (char*)&bbuf[tid * BSTR];
#if HAVE_ASYNC
        if (active) {
#pragma unroll
            for (int t = 0; t < 4; ++t)
                __builtin_amdgcn_global_load_async_to_lds_b128(
                    (gptr128)(g + 16 * t), (lptr128)(l + 16 * t), 0, 0);
        }
#else
        if (active) {
            uint4 wr[4];
            const uint4* gp = (const uint4*)g;
            uint4* lp = (uint4*)l;
#pragma unroll
            for (int t = 0; t < 4; ++t) wr[t] = gp[t];
#pragma unroll
            for (int t = 0; t < 4; ++t) lp[t] = wr[t];
        }
#endif
    };

    auto loadA = [&](Frag16& a, const _Float16* xbuf, int kofs) {
#pragma unroll
        for (int q = 0; q < 8; ++q) {
            int kb = kofs + ((q < 4) ? (q << 1) : (16 + ((q - 4) << 1))) + (khalf << 3);
            a.u[q] = *(const unsigned*)&xbuf[(rowTile * 16 + m) * XSTR + kb];
        }
    };
    auto loadAH = [&](Frag16& a, int kofs) {
#pragma unroll
        for (int q = 0; q < 8; ++q) {
            int kb = kofs + ((q < 4) ? (q << 1) : (16 + ((q - 4) << 1))) + (khalf << 3);
            a.u[q] = *(const unsigned*)&hs[(rowTile * 16 + m) * HSTR + kb];
        }
    };
    auto loadBF = [&](Frag16& b, const _Float16* bbuf, int ncol) {
        const _Float16* p = bbuf + ncol * BSTR + (khalf << 3);
#pragma unroll
        for (int q = 0; q < 8; ++q)
            b.u[q] = *(const unsigned*)(p + ((q < 4) ? (q << 1) : (16 + ((q - 4) << 1))));
    };

    // ---------------- Phase 1: H = SiLU(X @ W1 + b1) ----------------
    v8f acc[8] = {};

    stageW(w1t, (size_t)dinp, 0, bsbuf(0), true);
    loadX(0);
    storeX(xsbuf(0));
    wait_async();
    __syncthreads();

    for (int kc = 0; kc < nk1; ++kc) {
        const int cur = kc & 1;
        const bool hasNext = (kc + 1) < nk1;
        if (hasNext) {
            stageW(w1t, (size_t)dinp, kc + 1, bsbuf(cur ^ 1), true);  // async in flight
            loadX(kc + 1);                                           // global in flight
        }

        Frag16 a; loadA(a, xsbuf(cur), 0);
        Frag16 bf[2];
        loadBF(bf[0], bsbuf(cur), colHalf * 128 + m);
#pragma unroll
        for (int ct = 0; ct < 8; ++ct) {
            if (ct < 7)
                loadBF(bf[(ct + 1) & 1], bsbuf(cur), colHalf * 128 + (ct + 1) * 16 + m);
            acc[ct] = wmma16(a, bf[ct & 1], acc[ct]);
        }

        if (hasNext) storeX(xsbuf(cur ^ 1));
        wait_async();
        __syncthreads();
    }

    // bias + SiLU -> H (f16 in LDS)
#pragma unroll
    for (int ct = 0; ct < 8; ++ct) {
        int col = colHalf * 128 + ct * 16 + m;
        float bias = b1[col];
#pragma unroll
        for (int r = 0; r < 8; ++r) {
            int row = rowTile * 16 + (khalf << 3) + r;
            float x = acc[ct][r] + bias;
            hs[row * HSTR + col] = (_Float16)(x * fast_sigmoid(x));
        }
    }

    // ---------------- Phase 2: Y = H @ W2 + b2 ----------------
    constexpr int NCT2 = DOUT >> 5;       // tiles per wave (8 for 256, 2 for 64)

    v8f acc2[NCT2] = {};
    stageW(w2t, (size_t)256, 0, b2buf(0), tid < DOUT);
    wait_async();
    __syncthreads();   // also covers hs visibility

    for (int kc = 0; kc < 8; ++kc) {
        const int cur = kc & 1;
        const bool hasNext = kc < 7;
        if (hasNext) stageW(w2t, (size_t)256, kc + 1, b2buf(cur ^ 1), tid < DOUT);

        Frag16 a; loadAH(a, kc << 5);
        Frag16 bf[2];
        loadBF(bf[0], b2buf(cur), colHalf * (DOUT / 2) + m);
#pragma unroll
        for (int ct = 0; ct < NCT2; ++ct) {
            if (ct + 1 < NCT2)
                loadBF(bf[(ct + 1) & 1], b2buf(cur), colHalf * (DOUT / 2) + (ct + 1) * 16 + m);
            acc2[ct] = wmma16(a, bf[ct & 1], acc2[ct]);
        }

        wait_async();
        __syncthreads();
    }

    // spill Y tile (f32) to LDS for row-wise LayerNorm
    float* Ys = (float*)region;
    constexpr int YSTR = DOUT + 4;
#pragma unroll
    for (int ct = 0; ct < NCT2; ++ct) {
        int col = colHalf * (DOUT / 2) + ct * 16 + m;
#pragma unroll
        for (int r = 0; r < 8; ++r) {
            int row = rowTile * 16 + (khalf << 3) + r;
            Ys[row * YSTR + col] = acc2[ct][r];
        }
    }
    __syncthreads();

    // ---------------- Epilogue: bias2 + LN + residual + scatter ----------------
    {
        int r = tid >> 2, t = tid & 3;      // 4 lanes per row
        long R = R0 + r;
        if (R < M) {
            float* yr = Ys + r * YSTR;
            if (gw) {
                float s = 0.f;
                for (int c = t; c < DOUT; c += 4) s += yr[c] + b2[c];
                s += __shfl_xor(s, 1, 4);
                s += __shfl_xor(s, 2, 4);
                float mean = s / (float)DOUT;
                float vs = 0.f;
                for (int c = t; c < DOUT; c += 4) {
                    float d = yr[c] + b2[c] - mean; vs += d * d;
                }
                vs += __shfl_xor(vs, 1, 4);
                vs += __shfl_xor(vs, 2, 4);
                float rstd = rsqrtf(vs / (float)DOUT + 1e-5f);
                long dbase = aggp ? (long)dsti[R] * DOUT : 0;
                for (int c = t; c < DOUT; c += 4) {
                    float y = (yr[c] + b2[c] - mean) * rstd * gw[c] + bw[c];
                    if (resid) y += resid[R * (long)DOUT + c];
                    if (outp)  outp[R * (long)DOUT + c] = y;
                    if (aggp)  atomicAdd(&aggp[dbase + c], y);
                }
            } else {
                for (int c = t; c < DOUT; c += 4) {
                    float y = yr[c] + b2[c];
                    if (fstd) y = y * fstd[c] + fmean[c] + fgnf[R * 160 + 64 + c];
                    if (outp) outp[R * (long)DOUT + c] = y;
                }
            }
        }
    }
}

// ---------------------------------------------------------------------------
// Host side
// ---------------------------------------------------------------------------
extern "C" void kernel_launch(void* const* d_in, const int* in_sizes, int n_in,
                              void* d_out, int out_size, void* d_ws, size_t ws_size,
                              hipStream_t stream) {
    (void)in_sizes; (void)n_in; (void)out_size; (void)ws_size;

    const float* gnf   = (const float*)d_in[0];
    const float* mnf   = (const float*)d_in[1];
    const float* mef   = (const float*)d_in[2];
    const float* g2mef = (const float*)d_in[3];
    const float* m2gef = (const float*)d_in[4];
    const float* meanv = (const float*)d_in[5];
    const float* stdv  = (const float*)d_in[6];
    const int* g2m_s = (const int*)d_in[7];
    const int* g2m_d = (const int*)d_in[8];
    const int* m2m_s = (const int*)d_in[9];
    const int* m2m_d = (const int*)d_in[10];
    const int* m2g_s = (const int*)d_in[11];
    const int* m2g_d = (const int*)d_in[12];

    // params: jax pytree leaf order (sorted dict keys; lists in order)
    auto P = [&](int i) { return (const float*)d_in[i]; };
    struct Mlp { const float *b1,*b2,*beta,*g,*w1,*w2; int din, dout; };
    auto mk = [&](int base, int din, int dout) {
        Mlp mm; mm.b1 = P(base); mm.b2 = P(base+1); mm.beta = P(base+2); mm.g = P(base+3);
        mm.w1 = P(base+4); mm.w2 = P(base+5); mm.din = din; mm.dout = dout; return mm;
    };
    Mlp dec_edge = mk(13, 768, 256);
    Mlp dec_node = mk(19, 512, 256);
    Mlp emb_eg2m = mk(25, 4, 256);
    Mlp emb_em   = mk(31, 4, 256);
    Mlp emb_em2g = mk(37, 4, 256);
    Mlp emb_vg   = mk(43, 160, 256);
    Mlp emb_vm   = mk(49, 3, 256);
    Mlp fin; fin.b1 = P(55); fin.b2 = P(56); fin.beta = nullptr; fin.g = nullptr;
    fin.w1 = P(57); fin.w2 = P(58); fin.din = 256; fin.dout = 64;
    Mlp g2m_edge = mk(59, 768, 256);
    Mlp g2m_grid = mk(65, 256, 256);
    Mlp g2m_node = mk(71, 512, 256);
    Mlp proc_e[4], proc_n[4];
    for (int s = 0; s < 4; ++s) {
        proc_e[s] = mk(77 + 12 * s,     768, 256);
        proc_n[s] = mk(77 + 12 * s + 6, 512, 256);
    }

    // workspace bump allocator
    char* wsb = (char*)d_ws;
    size_t off = 0;
    auto alloc = [&](size_t bytes) -> void* {
        off = (off + 255) & ~(size_t)255;
        void* p = wsb + off; off += bytes; return p;
    };

    // convert weights -> transposed, K-padded f16
    struct WH { _Float16 *w1t, *w2t; };
    auto cvt = [&](const Mlp& mp) -> WH {
        WH h;
        int dinp = (mp.din + 31) & ~31;
        h.w1t = (_Float16*)alloc((size_t)256 * dinp * 2);
        h.w2t = (_Float16*)alloc((size_t)mp.dout * 256 * 2);
        int n1 = 256 * dinp;
        k_cvt_t<<<(n1 + 255) / 256, 256, 0, stream>>>(mp.w1, h.w1t, mp.din, 256, dinp);
        int n2 = mp.dout * 256;
        k_cvt_t<<<(n2 + 255) / 256, 256, 0, stream>>>(mp.w2, h.w2t, 256, mp.dout, 256);
        return h;
    };
    WH h_vg = cvt(emb_vg), h_vm = cvt(emb_vm), h_em = cvt(emb_em);
    WH h_eg2m = cvt(emb_eg2m), h_em2g = cvt(emb_em2g);
    WH h_ge = cvt(g2m_edge), h_gn = cvt(g2m_node), h_gg = cvt(g2m_grid);
    WH h_de = cvt(dec_edge), h_dn = cvt(dec_node), h_fin = cvt(fin);
    WH h_pe[4], h_pn[4];
    for (int s = 0; s < 4; ++s) { h_pe[s] = cvt(proc_e[s]); h_pn[s] = cvt(proc_n[s]); }

    // activation buffers
    const int NG = 32768, NM = 10242, EM = 81920, EG = 98304;
    float* vg   = (float*)alloc((size_t)NG * 256 * 4);
    float* vm   = (float*)alloc((size_t)NM * 256 * 4);
    float* em   = (float*)alloc((size_t)EM * 256 * 4);
    float* E1   = (float*)alloc((size_t)EG * 256 * 4);   // eg2m, later em2g
    float* aggM = (float*)alloc((size_t)NM * 256 * 4);
    float* aggG = (float*)alloc((size_t)NG * 256 * 4);

    auto run = [&](int M, const Mlp& mp, const WH& wh,
                   const float* a0, const int* ia0, int w0,
                   const float* a1, const int* ia1,
                   const float* a2, const int* ia2,
                   const float* resid, float* outp,
                   float* aggp, const int* dsti,
                   const float* fstd, const float* fmn, const float* fg) {
        dim3 grd((M + BM - 1) / BM), blk(256);
        if (mp.dout == 256)
            k_mlp<256><<<grd, blk, 0, stream>>>(M, mp.din,
                a0, ia0, w0, a1, ia1, a2, ia2,
                wh.w1t, mp.b1, wh.w2t, mp.b2, mp.g, mp.beta,
                resid, outp, aggp, dsti, fstd, fmn, fg);
        else
            k_mlp<64><<<grd, blk, 0, stream>>>(M, mp.din,
                a0, ia0, w0, a1, ia1, a2, ia2,
                wh.w1t, mp.b1, wh.w2t, mp.b2, mp.g, mp.beta,
                resid, outp, aggp, dsti, fstd, fmn, fg);
    };

    // ---- Encoder embeddings ----
    run(NG, emb_vg, h_vg, gnf, nullptr, 160, nullptr, nullptr, nullptr, nullptr,
        nullptr, vg, nullptr, nullptr, nullptr, nullptr, nullptr);
    run(NM, emb_vm, h_vm, mnf, nullptr, 3, nullptr, nullptr, nullptr, nullptr,
        nullptr, vm, nullptr, nullptr, nullptr, nullptr, nullptr);
    run(EM, emb_em, h_em, mef, nullptr, 4, nullptr, nullptr, nullptr, nullptr,
        nullptr, em, nullptr, nullptr, nullptr, nullptr, nullptr);
    run(EG, emb_eg2m, h_eg2m, g2mef, nullptr, 4, nullptr, nullptr, nullptr, nullptr,
        nullptr, E1, nullptr, nullptr, nullptr, nullptr, nullptr);

    // ---- grid -> mesh ----
    (void)hipMemsetAsync(aggM, 0, (size_t)NM * 256 * 4, stream);
    run(EG, g2m_edge, h_ge, E1, nullptr, 256, vg, g2m_s, vm, g2m_d,
        nullptr, nullptr, aggM, g2m_d, nullptr, nullptr, nullptr);
    run(NM, g2m_node, h_gn, vm, nullptr, 256, aggM, nullptr, nullptr, nullptr,
        vm, vm, nullptr, nullptr, nullptr, nullptr, nullptr);
    run(NG, g2m_grid, h_gg, vg, nullptr, 256, nullptr, nullptr, nullptr, nullptr,
        vg, vg, nullptr, nullptr, nullptr, nullptr, nullptr);

    // ---- processor (4 steps) ----
    for (int s = 0; s < 4; ++s) {
        (void)hipMemsetAsync(aggM, 0, (size_t)NM * 256 * 4, stream);
        run(EM, proc_e[s], h_pe[s], em, nullptr, 256, vm, m2m_s, vm, m2m_d,
            em, em, aggM, m2m_d, nullptr, nullptr, nullptr);
        run(NM, proc_n[s], h_pn[s], vm, nullptr, 256, aggM, nullptr, nullptr, nullptr,
            vm, vm, nullptr, nullptr, nullptr, nullptr, nullptr);
    }

    // ---- decoder ----
    run(EG, emb_em2g, h_em2g, m2gef, nullptr, 4, nullptr, nullptr, nullptr, nullptr,
        nullptr, E1, nullptr, nullptr, nullptr, nullptr, nullptr);
    (void)hipMemsetAsync(aggG, 0, (size_t)NG * 256 * 4, stream);
    run(EG, dec_edge, h_de, E1, nullptr, 256, vm, m2g_s, vg, m2g_d,
        nullptr, nullptr, aggG, m2g_d, nullptr, nullptr, nullptr);
    run(NG, dec_node, h_dn, vg, nullptr, 256, aggG, nullptr, nullptr, nullptr,
        vg, vg, nullptr, nullptr, nullptr, nullptr, nullptr);

    // ---- output head: y*std + mean + gnf[:,64:128] ----
    run(NG, fin, h_fin, vg, nullptr, 256, nullptr, nullptr, nullptr, nullptr,
        nullptr, (float*)d_out, nullptr, nullptr, stdv, meanv, gnf);
}
